// OpusDecoder_34445637714414
// MI455X (gfx1250) — matrix operand order repaired and checked by reading, not compiled, not run on hardware
//
#include <hip/hip_runtime.h>

// ---------------------------------------------------------------------------
// 2-layer LLaMA-style decoder forward for MI455X (gfx1250, wave32, WMMA).
// All GEMMs run as bf16 x bf16 -> f32 via v_wmma_f32_16x16x32_bf16, with
// double-buffered GLOBAL_LOAD_ASYNC_TO_LDS_B128 tile staging (ASYNCcnt).
// Workspace requirement: ~791 MB carved from d_ws.
// ---------------------------------------------------------------------------

typedef __attribute__((ext_vector_type(16))) __bf16 v16bf;
typedef __attribute__((ext_vector_type(8)))  float  v8f;
typedef __attribute__((vector_size(16)))     int    v4i;   // 128-bit element

#define AS3 __attribute__((address_space(3)))

#if defined(__has_builtin)
#if __has_builtin(__builtin_amdgcn_global_load_async_to_lds_b128)
#define USE_ASYNC_LDS 1
#endif
#endif
#ifndef USE_ASYNC_LDS
#define USE_ASYNC_LDS 0
#endif

__device__ __forceinline__ void wait_async0() {
#if USE_ASYNC_LDS
#if __has_builtin(__builtin_amdgcn_s_wait_asynccnt)
  __builtin_amdgcn_s_wait_asynccnt(0);
#else
  asm volatile("s_wait_asynccnt 0x0" ::: "memory");
#endif
#endif
}

#define DMODEL 2048
#define NHEAD  16
#define HEADD  128
#define NLAYER 2
#define HIDDEN 8192
#define BATCHN 2
#define SEQ    1024
#define NTOK   (BATCHN * SEQ)
#define VOCAB  32000

__device__ __forceinline__ unsigned short f2bf(float f) {
  unsigned int u = __float_as_uint(f);
  u += 0x7FFFu + ((u >> 16) & 1u);           // round-to-nearest-even
  return (unsigned short)(u >> 16);
}

// ---------------------------------------------------------------------------
// Generic f32 -> bf16 conversion (grid-stride)
// ---------------------------------------------------------------------------
__global__ __launch_bounds__(256) void cvt_f32_bf16(
    const float* __restrict__ src, unsigned short* __restrict__ dst, long long n) {
  long long i = (long long)blockIdx.x * blockDim.x + threadIdx.x;
  long long stride = (long long)gridDim.x * blockDim.x;
  for (; i < n; i += stride) dst[i] = f2bf(src[i]);
}

// ---------------------------------------------------------------------------
// Embedding gather: x[row, :] = embed[token[row], :]   (f32 residual stream)
// ---------------------------------------------------------------------------
__global__ __launch_bounds__(256) void gather_embed(
    const int* __restrict__ tok, const float* __restrict__ embed,
    float* __restrict__ x) {
  int row = blockIdx.x;
  const float* e = embed + (size_t)tok[row] * DMODEL;
  float* xr = x + (size_t)row * DMODEL;
  for (int c = threadIdx.x; c < DMODEL; c += 256) xr[c] = e[c];
}

// ---------------------------------------------------------------------------
// RMSNorm: out_bf16[row,:] = x * rsqrt(mean(x^2)+eps) * w
// ---------------------------------------------------------------------------
__global__ __launch_bounds__(256) void rmsnorm_kernel(
    const float* __restrict__ x, const float* __restrict__ w,
    unsigned short* __restrict__ out, int ncols) {
  int row = blockIdx.x;
  const float* xr = x + (size_t)row * ncols;
  float ss = 0.f;
  for (int c = threadIdx.x; c < ncols; c += 256) { float v = xr[c]; ss += v * v; }
  __shared__ float red[256];
  red[threadIdx.x] = ss; __syncthreads();
  for (int t = 128; t > 0; t >>= 1) {
    if (threadIdx.x < t) red[threadIdx.x] += red[threadIdx.x + t];
    __syncthreads();
  }
  float rs = rsqrtf(red[0] / (float)ncols + 1e-6f);
  unsigned short* orow = out + (size_t)row * ncols;
  for (int c = threadIdx.x; c < ncols; c += 256) orow[c] = f2bf(xr[c] * rs * w[c]);
}

// ---------------------------------------------------------------------------
// RoPE on q (f32 in [NTOK, D]) -> bf16 [NTOK, D]
// ---------------------------------------------------------------------------
__global__ __launch_bounds__(256) void rope_q_kernel(
    const float* __restrict__ qf, unsigned short* __restrict__ qb) {
  int idx = blockIdx.x * 256 + threadIdx.x;     // NTOK*NHEAD*64 threads
  int d = idx & 63;
  int h = (idx >> 6) & (NHEAD - 1);
  int m = idx >> 10;                             // token index
  int s = m & (SEQ - 1);
  float inv = __powf(10000.0f, -(float)(2 * d) / (float)HEADD);
  float sn, cs; __sincosf((float)s * inv, &sn, &cs);
  size_t base = (size_t)m * DMODEL + h * HEADD + d;
  float q1 = qf[base], q2 = qf[base + 64];
  qb[base]      = f2bf(q1 * cs - q2 * sn);
  qb[base + 64] = f2bf(q2 * cs + q1 * sn);
}

// ---------------------------------------------------------------------------
// RoPE on k + extract v from kv (f32 [NTOK, 256]) -> bf16 k,v [NTOK, 128]
// ---------------------------------------------------------------------------
__global__ __launch_bounds__(256) void rope_kv_kernel(
    const float* __restrict__ kv, unsigned short* __restrict__ kb,
    unsigned short* __restrict__ vb) {
  int idx = blockIdx.x * 256 + threadIdx.x;     // NTOK*64 threads
  int d = idx & 63;
  int m = idx >> 6;
  int s = m & (SEQ - 1);
  float inv = __powf(10000.0f, -(float)(2 * d) / (float)HEADD);
  float sn, cs; __sincosf((float)s * inv, &sn, &cs);
  const float* r = kv + (size_t)m * (2 * HEADD);
  float k1 = r[d], k2 = r[d + 64];
  size_t o = (size_t)m * HEADD + d;
  kb[o]      = f2bf(k1 * cs - k2 * sn);
  kb[o + 64] = f2bf(k2 * cs + k1 * sn);
  vb[o]      = f2bf(r[HEADD + d]);
  vb[o + 64] = f2bf(r[HEADD + 64 + d]);
}

// ---------------------------------------------------------------------------
// ALiBi + causal softmax over a row of scores; emits bf16 probabilities.
// row = (b*NHEAD + h)*SEQ + s
// ---------------------------------------------------------------------------
__global__ __launch_bounds__(256) void softmax_alibi(
    const float* __restrict__ sc, unsigned short* __restrict__ pr) {
  int row = blockIdx.x;
  int s = row & (SEQ - 1);
  int h = (row >> 10) & (NHEAD - 1);
  const float scale = 0.088388347648318447f;     // 1/sqrt(128)
  float slope = exp2f(-8.0f * (float)(h + 1) / (float)NHEAD);
  const float* r = sc + (size_t)row * SEQ;
  __shared__ float red[256];
  float vals[4];
  float mx = -3.4e38f;
#pragma unroll
  for (int i = 0; i < 4; ++i) {
    int T = threadIdx.x + i * 256;
    float v = (T <= s) ? (r[T] * scale + slope * (float)(T - s)) : -3.4e38f;
    vals[i] = v;
    mx = fmaxf(mx, v);
  }
  red[threadIdx.x] = mx; __syncthreads();
  for (int t = 128; t > 0; t >>= 1) {
    if (threadIdx.x < t) red[threadIdx.x] = fmaxf(red[threadIdx.x], red[threadIdx.x + t]);
    __syncthreads();
  }
  mx = red[0]; __syncthreads();
  float sum = 0.f;
#pragma unroll
  for (int i = 0; i < 4; ++i) { vals[i] = __expf(vals[i] - mx); sum += vals[i]; }
  red[threadIdx.x] = sum; __syncthreads();
  for (int t = 128; t > 0; t >>= 1) {
    if (threadIdx.x < t) red[threadIdx.x] += red[threadIdx.x + t];
    __syncthreads();
  }
  float invs = 1.0f / red[0];
  unsigned short* prow = pr + (size_t)row * SEQ;
#pragma unroll
  for (int i = 0; i < 4; ++i) prow[threadIdx.x + i * 256] = f2bf(vals[i] * invs);
}

// ---------------------------------------------------------------------------
// SwiGLU: out_bf16 = silu(g1) * g2
// ---------------------------------------------------------------------------
__global__ __launch_bounds__(256) void swiglu_kernel(
    const float* __restrict__ g1, const float* __restrict__ g2,
    unsigned short* __restrict__ out, long long n) {
  long long i = (long long)blockIdx.x * 256 + threadIdx.x;
  long long stride = (long long)gridDim.x * 256;
  for (; i < n; i += stride) {
    float g = g1[i];
    out[i] = f2bf((g / (1.0f + __expf(-g))) * g2[i]);
  }
}

// ---------------------------------------------------------------------------
// Tiled WMMA GEMM (bf16 x bf16 -> f32):
//   C[M,N] (+)= A[M,K] @ op(B)      op(B) = B[N,K]^T if TRANSB else B[K,N]
// Block tile 128x64, 8 waves, each wave computes 32x32 via 2x2 WMMA frags.
// K stepped by 32. Double-buffered LDS; tiles staged with
// GLOBAL_LOAD_ASYNC_TO_LDS_B128 (ASYNCcnt) when available, so the next tile
// streams into the alternate buffer while WMMAs consume the current one.
// One barrier per K-step: readers of buffer b finish before the top barrier
// of the iteration that re-issues writes into b.
// MODE 0: C = acc ; MODE 1: C += acc  (residual accumulation)
// All problem dims here divide tile sizes evenly, so no bounds checks.
// ---------------------------------------------------------------------------
template <int MODE, int TRANSB>
__global__ __launch_bounds__(256) void gemm_wmma(
    const unsigned short* __restrict__ A, const unsigned short* __restrict__ B,
    float* __restrict__ C, int M, int N, int K, int lda, int ldb, int ldc,
    int innerCount, long long sAo, long long sAi, long long sBo, long long sBi,
    long long sCo, long long sCi) {
  __shared__ unsigned short As[2][128 * 40];   // [m][k] tile, padded stride 40
  __shared__ unsigned short Bs[2][64 * 40];    // [n][k] tile (B staged transposed)

  const int z = blockIdx.z;
  const int zo = z / innerCount, zi = z % innerCount;
  A += zo * sAo + zi * sAi;
  B += zo * sBo + zi * sBi;
  C += zo * sCo + zi * sCi;

  const int m0 = blockIdx.y * 128;
  const int n0 = blockIdx.x * 64;
  const int tid = threadIdx.x;
  const int lane = tid & 31;
  const int wid = tid >> 5;
  const int wm = (wid >> 1) * 32;   // wave row offset: 0..96
  const int wn = (wid & 1) * 32;    // wave col offset: 0,32
  const int l15 = lane & 15;
  const int kh = lane >> 4;         // K-half select per ISA A/B VGPR layout

  v8f acc[2][2] = {};

  const int arow = tid >> 1, ahalf = (tid & 1) * 16;   // A: 2 thr/row, 16 bf16 ea
  const int brow = tid >> 2, bq = (tid & 3) * 8;       // B transB: 4 thr/row
  const int bk = tid >> 3, bj = (tid & 7) * 8;         // B !transB: scatter

  auto load_tiles = [&](int kk, int buf) {
#if USE_ASYNC_LDS
    {
      const unsigned short* g = A + (size_t)(m0 + arow) * lda + kk + ahalf;
      __builtin_amdgcn_global_load_async_to_lds_b128(
          (v4i*)g, (AS3 v4i*)&As[buf][arow * 40 + ahalf], 0, 0);
      __builtin_amdgcn_global_load_async_to_lds_b128(
          (v4i*)(g + 8), (AS3 v4i*)&As[buf][arow * 40 + ahalf + 8], 0, 0);
    }
    if (TRANSB) {
      const unsigned short* g = B + (size_t)(n0 + brow) * ldb + kk + bq;
      __builtin_amdgcn_global_load_async_to_lds_b128(
          (v4i*)g, (AS3 v4i*)&Bs[buf][brow * 40 + bq], 0, 0);
    } else {
      union { uint4 u; unsigned short s[8]; } t;
      t.u = *(const uint4*)(B + (size_t)(kk + bk) * ldb + n0 + bj);
#pragma unroll
      for (int j = 0; j < 8; ++j) Bs[buf][(bj + j) * 40 + bk] = t.s[j];
    }
#else
    {
      const uint4* g = (const uint4*)(A + (size_t)(m0 + arow) * lda + kk + ahalf);
      *(uint4*)&As[buf][arow * 40 + ahalf] = g[0];
      *(uint4*)&As[buf][arow * 40 + ahalf + 8] = g[1];
    }
    if (TRANSB) {
      const uint4* g = (const uint4*)(B + (size_t)(n0 + brow) * ldb + kk + bq);
      *(uint4*)&Bs[buf][brow * 40 + bq] = g[0];
    } else {
      union { uint4 u; unsigned short s[8]; } t;
      t.u = *(const uint4*)(B + (size_t)(kk + bk) * ldb + n0 + bj);
#pragma unroll
      for (int j = 0; j < 8; ++j) Bs[buf][(bj + j) * 40 + bk] = t.s[j];
    }
#endif
  };

  load_tiles(0, 0);                    // prologue: stage first tile into buf 0

  int buf = 0;
  for (int kk = 0; kk < K; kk += 32, buf ^= 1) {
    wait_async0();                     // this wave's async writes have landed
    __syncthreads();                   // all waves' writes visible; buf^1 free
    if (kk + 32 < K) load_tiles(kk + 32, buf ^ 1);   // prefetch next tile

    union Frag { v16bf v; uint4 u[2]; } a[2], b[2];
#pragma unroll
    for (int tm = 0; tm < 2; ++tm) {
      int r = wm + tm * 16 + l15;
      a[tm].u[0] = *(const uint4*)&As[buf][r * 40 + kh * 8];       // K 0-7 / 8-15
      a[tm].u[1] = *(const uint4*)&As[buf][r * 40 + 16 + kh * 8];  // K 16-23 / 24-31
    }
#pragma unroll
    for (int tn = 0; tn < 2; ++tn) {
      int c = wn + tn * 16 + l15;
      b[tn].u[0] = *(const uint4*)&Bs[buf][c * 40 + kh * 8];       // K 0-7 / 8-15
      b[tn].u[1] = *(const uint4*)&Bs[buf][c * 40 + 16 + kh * 8];  // K 16-23 / 24-31
    }
#pragma unroll
    for (int tm = 0; tm < 2; ++tm)
#pragma unroll
      for (int tn = 0; tn < 2; ++tn)
        acc[tm][tn] = __builtin_amdgcn_wmma_f32_16x16x32_bf16(
            false, a[tm].v, false, b[tn].v, (short)0, acc[tm][tn], false, false);
  }

  // C/D layout: VGPR r, lanes 0-15 -> M=r, N=lane ; lanes 16-31 -> M=r+8
  const int rbase = m0 + wm + ((lane >= 16) ? 8 : 0);
  const int cbase = n0 + wn + l15;
#pragma unroll
  for (int tm = 0; tm < 2; ++tm)
#pragma unroll
    for (int tn = 0; tn < 2; ++tn)
#pragma unroll
      for (int r = 0; r < 8; ++r) {
        size_t idx = (size_t)(rbase + tm * 16 + r) * ldc + cbase + tn * 16;
        if (MODE == 0) C[idx] = acc[tm][tn][r];
        else           C[idx] += acc[tm][tn][r];
      }
}

// ---------------------------------------------------------------------------
// Host orchestration
// ---------------------------------------------------------------------------
extern "C" void kernel_launch(void* const* d_in, const int* in_sizes, int n_in,
                              void* d_out, int out_size, void* d_ws, size_t ws_size,
                              hipStream_t stream) {
  (void)in_sizes; (void)n_in; (void)out_size; (void)ws_size;
  const int*   tok   = (const int*)d_in[0];
  const float* embed = (const float*)d_in[1];
  const float* ln1   = (const float*)d_in[2];
  const float* Wq    = (const float*)d_in[3];
  const float* Wkv   = (const float*)d_in[4];
  const float* Wo    = (const float*)d_in[5];
  const float* ln2   = (const float*)d_in[6];
  const float* W1    = (const float*)d_in[7];
  const float* W2    = (const float*)d_in[8];
  const float* W3    = (const float*)d_in[9];
  const float* lnf   = (const float*)d_in[10];
  float* out = (float*)d_out;

  char* p = (char*)d_ws;
  auto alloc = [&](size_t bytes) -> char* {
    char* r = p; p += (bytes + 255) & ~(size_t)255; return r;
  };
  typedef unsigned short u16;
  u16* embedB = (u16*)alloc((size_t)VOCAB * DMODEL * 2);
  u16* WqB    = (u16*)alloc((size_t)NLAYER * DMODEL * DMODEL * 2);
  u16* WkvB   = (u16*)alloc((size_t)NLAYER * 2 * HEADD * DMODEL * 2);
  u16* WoB    = (u16*)alloc((size_t)NLAYER * DMODEL * DMODEL * 2);
  u16* W1B    = (u16*)alloc((size_t)NLAYER * HIDDEN * DMODEL * 2);
  u16* W2B    = (u16*)alloc((size_t)NLAYER * HIDDEN * DMODEL * 2);
  u16* W3B    = (u16*)alloc((size_t)NLAYER * DMODEL * HIDDEN * 2);
  float* xF   = (float*)alloc((size_t)NTOK * DMODEL * 4);
  u16*   hB   = (u16*)alloc((size_t)NTOK * DMODEL * 2);   // h / h2 / o_bf16 / xn
  float* qF   = (float*)alloc((size_t)NTOK * DMODEL * 4); // q pre-rope, later o
  float* kvF  = (float*)alloc((size_t)NTOK * 2 * HEADD * 4);
  u16*   qB   = (u16*)alloc((size_t)NTOK * DMODEL * 2);
  u16*   kB   = (u16*)alloc((size_t)NTOK * HEADD * 2);
  u16*   vB   = (u16*)alloc((size_t)NTOK * HEADD * 2);
  float* scF  = (float*)alloc((size_t)BATCHN * NHEAD * SEQ * SEQ * 4);
  u16*   prB  = (u16*)alloc((size_t)BATCHN * NHEAD * SEQ * SEQ * 2);
  float* g1F  = (float*)alloc((size_t)NTOK * HIDDEN * 4);
  float* g2F  = (float*)alloc((size_t)NTOK * HIDDEN * 4);
  u16*   gB   = (u16*)alloc((size_t)NTOK * HIDDEN * 2);

  auto cgrid = [](long long n) {
    long long b = (n + 255) / 256; return (unsigned)(b > 32768 ? 32768 : b);
  };
  auto cvt = [&](const float* s, u16* d, long long n) {
    cvt_f32_bf16<<<cgrid(n), 256, 0, stream>>>(s, d, n);
  };

  // --- weight down-conversion to bf16 (fresh every call; deterministic) ---
  cvt(embed, embedB, (long long)VOCAB * DMODEL);
  cvt(Wq,  WqB,  (long long)NLAYER * DMODEL * DMODEL);
  cvt(Wkv, WkvB, (long long)NLAYER * 2 * HEADD * DMODEL);
  cvt(Wo,  WoB,  (long long)NLAYER * DMODEL * DMODEL);
  cvt(W1,  W1B,  (long long)NLAYER * HIDDEN * DMODEL);
  cvt(W2,  W2B,  (long long)NLAYER * HIDDEN * DMODEL);
  cvt(W3,  W3B,  (long long)NLAYER * DMODEL * HIDDEN);

  gather_embed<<<NTOK, 256, 0, stream>>>(tok, embed, xF);

  for (int l = 0; l < NLAYER; ++l) {
    const u16* WqL  = WqB  + (size_t)l * DMODEL * DMODEL;
    const u16* WkvL = WkvB + (size_t)l * 2 * HEADD * DMODEL;
    const u16* WoL  = WoB  + (size_t)l * DMODEL * DMODEL;
    const u16* W1L  = W1B  + (size_t)l * HIDDEN * DMODEL;
    const u16* W2L  = W2B  + (size_t)l * HIDDEN * DMODEL;
    const u16* W3L  = W3B  + (size_t)l * DMODEL * HIDDEN;

    // h = rmsnorm(x, ln1[l])
    rmsnorm_kernel<<<NTOK, 256, 0, stream>>>(xF, ln1 + (size_t)l * DMODEL, hB, DMODEL);

    // q = h @ Wq^T   [2048 x 2048 x 2048]
    gemm_wmma<0, 1><<<dim3(DMODEL / 64, NTOK / 128, 1), 256, 0, stream>>>(
        hB, WqL, qF, NTOK, DMODEL, DMODEL, DMODEL, DMODEL, DMODEL,
        1, 0, 0, 0, 0, 0, 0);
    // kv = h @ Wkv^T [2048 x 256 x 2048]
    gemm_wmma<0, 1><<<dim3(2 * HEADD / 64, NTOK / 128, 1), 256, 0, stream>>>(
        hB, WkvL, kvF, NTOK, 2 * HEADD, DMODEL, DMODEL, DMODEL, 2 * HEADD,
        1, 0, 0, 0, 0, 0, 0);

    rope_q_kernel<<<(NTOK * NHEAD * 64) / 256, 256, 0, stream>>>(qF, qB);
    rope_kv_kernel<<<(NTOK * 64) / 256, 256, 0, stream>>>(kvF, kB, vB);

    // scores[b,h] = q[b,:,h,:] @ k[b]^T   [1024 x 1024 x 128] x 32 batches
    gemm_wmma<0, 1><<<dim3(SEQ / 64, SEQ / 128, BATCHN * NHEAD), 256, 0, stream>>>(
        qB, kB, scF, SEQ, SEQ, HEADD, DMODEL, HEADD, SEQ, NHEAD,
        (long long)SEQ * DMODEL, (long long)HEADD,
        (long long)SEQ * HEADD, 0,
        (long long)NHEAD * SEQ * SEQ, (long long)SEQ * SEQ);

    softmax_alibi<<<BATCHN * NHEAD * SEQ, 256, 0, stream>>>(scF, prB);

    // o[b,:,h,:] = probs[b,h] @ v[b]   [1024 x 128 x 1024] x 32 batches
    gemm_wmma<0, 0><<<dim3(HEADD / 64, SEQ / 128, BATCHN * NHEAD), 256, 0, stream>>>(
        prB, vB, qF, SEQ, HEADD, SEQ, SEQ, HEADD, DMODEL, NHEAD,
        (long long)NHEAD * SEQ * SEQ, (long long)SEQ * SEQ,
        (long long)SEQ * HEADD, 0,
        (long long)SEQ * DMODEL, (long long)HEADD);

    // o -> bf16 (reuse hB), then x += o @ Wo^T
    cvt(qF, hB, (long long)NTOK * DMODEL);
    gemm_wmma<1, 1><<<dim3(DMODEL / 64, NTOK / 128, 1), 256, 0, stream>>>(
        hB, WoL, xF, NTOK, DMODEL, DMODEL, DMODEL, DMODEL, DMODEL,
        1, 0, 0, 0, 0, 0, 0);

    // MLP
    rmsnorm_kernel<<<NTOK, 256, 0, stream>>>(xF, ln2 + (size_t)l * DMODEL, hB, DMODEL);
    gemm_wmma<0, 1><<<dim3(HIDDEN / 64, NTOK / 128, 1), 256, 0, stream>>>(
        hB, W1L, g1F, NTOK, HIDDEN, DMODEL, DMODEL, DMODEL, HIDDEN,
        1, 0, 0, 0, 0, 0, 0);
    gemm_wmma<0, 1><<<dim3(HIDDEN / 64, NTOK / 128, 1), 256, 0, stream>>>(
        hB, W2L, g2F, NTOK, HIDDEN, DMODEL, DMODEL, DMODEL, HIDDEN,
        1, 0, 0, 0, 0, 0, 0);
    swiglu_kernel<<<cgrid((long long)NTOK * HIDDEN), 256, 0, stream>>>(
        g1F, g2F, gB, (long long)NTOK * HIDDEN);
    gemm_wmma<1, 1><<<dim3(DMODEL / 64, NTOK / 128, 1), 256, 0, stream>>>(
        gB, W3L, xF, NTOK, DMODEL, HIDDEN, HIDDEN, HIDDEN, DMODEL,
        1, 0, 0, 0, 0, 0, 0);
  }

  // Final norm + logits = xn @ embed^T  [2048 x 32000 x 2048]
  rmsnorm_kernel<<<NTOK, 256, 0, stream>>>(xF, lnf, hB, DMODEL);
  gemm_wmma<0, 1><<<dim3(VOCAB / 64, NTOK / 128, 1), 256, 0, stream>>>(
      hB, embedB, out, NTOK, VOCAB, DMODEL, DMODEL, DMODEL, VOCAB,
      1, 0, 0, 0, 0, 0, 0);
}
